// MultiHeadAttention_61529701483253
// MI455X (gfx1250) — matrix operand order repaired and checked by reading, not compiled, not run on hardware
//
#include <hip/hip_runtime.h>
#include <hip/hip_bf16.h>
#include <cstdint>

#define HEADS 16
#define HDIM  64
#define EMBED 1024
#define BATCH 4
#define SEQ   2048

typedef __attribute__((ext_vector_type(16))) _Float16 v16h;
typedef __attribute__((ext_vector_type(8)))  _Float16 v8h;
typedef __attribute__((ext_vector_type(8)))  float    v8f;
typedef __attribute__((ext_vector_type(4)))  uint32_t u32x4;
typedef __attribute__((ext_vector_type(8)))  uint32_t u32x8;

static __device__ __forceinline__ v8f wmma_f32_f16(v16h a, v16h b, v8f c) {
  // (neg_a, A, neg_b, B, c_mod, C, reuse_a, reuse_b)
  return __builtin_amdgcn_wmma_f32_16x16x32_f16(false, a, false, b, (short)0, c, false, false);
}

// ---- fragment loaders -------------------------------------------------------
// A fragment (16x32 f16) from row-major f16 matrix; lane supplies row m=l16.
// Lanes 0-15 hold K = kb+[0..7],kb+[16..23]; lanes 16-31 hold kb+[8..15],kb+[24..31].
static __device__ __forceinline__ v16h load_a_f16(const _Float16* base, int stride,
                                                  int m, int kb, int half) {
  const _Float16* p = base + (size_t)m * stride + kb + half * 8;
  v8h lo = *(const v8h*)p;
  v8h hi = *(const v8h*)(p + 16);
  return __builtin_shufflevector(lo, hi, 0,1,2,3,4,5,6,7,8,9,10,11,12,13,14,15);
}

// Same A fragment but from f32 source with on-the-fly convert.
static __device__ __forceinline__ v16h load_a_f32(const float* base, int stride,
                                                  int m, int kb, int half) {
  const float* p = base + (size_t)m * stride + kb + half * 8;
  v16h r;
#pragma unroll
  for (int i = 0; i < 8; ++i) r[i] = (_Float16)p[i];
#pragma unroll
  for (int i = 0; i < 8; ++i) r[8 + i] = (_Float16)p[16 + i];
  return r;
}

// B fragment (32x16 f16). Memory holds B^T row-major: bt[n][k].
// Lanes 0-15: K = kb+[0..7],kb+[16..23]; lanes 16-31: kb+[8..15],kb+[24..31] (column n0+l16).
static __device__ __forceinline__ v16h load_b_t(const _Float16* bt, int stride,
                                                int n0, int kb, int half, int l16) {
  const _Float16* p = bt + (size_t)(n0 + l16) * stride + kb + half * 8;
  v8h lo = *(const v8h*)p;
  v8h hi = *(const v8h*)(p + 16);
  return __builtin_shufflevector(lo, hi, 0,1,2,3,4,5,6,7,8,9,10,11,12,13,14,15);
}

// 16-lane butterfly reductions (masks 1..8 never cross the 16-lane halves).
static __device__ __forceinline__ float rowred_max(float x) {
#pragma unroll
  for (int m = 1; m < 16; m <<= 1) x = fmaxf(x, __shfl_xor(x, m, 32));
  return x;
}
static __device__ __forceinline__ float rowred_sum(float x) {
#pragma unroll
  for (int m = 1; m < 16; m <<= 1) x += __shfl_xor(x, m, 32);
  return x;
}

// ---- Tensor Data Mover: 2D tile global -> LDS (ISA 08_async_tensor 8.3/8.4) --
// data_size = 4B elements. Dimensions/strides in dword units.
// One issue per wave; tracked with TENSORcnt.
static __device__ __forceinline__ void tdm_load_2d(const void* lds_ptr, const void* gptr,
                                                   uint32_t tensor_d0, uint32_t tensor_d1,
                                                   uint32_t tile_d0, uint32_t tile_d1,
                                                   uint64_t stride0) {
  // generic->LDS: low 32 bits of a generic LDS pointer are the LDS byte offset
  uint32_t lds_off = (uint32_t)(uintptr_t)lds_ptr;
  uint64_t ga = (uint64_t)(uintptr_t)gptr;
  u32x4 g0;
  g0[0] = 1u;                                                   // count=1, user mode
  g0[1] = lds_off;                                              // lds_addr
  g0[2] = (uint32_t)ga;                                         // global_addr[31:0]
  g0[3] = (uint32_t)((ga >> 32) & 0x01FFFFFFu) | (2u << 30);    // addr[56:32] | type=2
  u32x8 g1;
  g1[0] = (2u << 16);                                           // data_size=4B, no flags
  g1[1] = (tensor_d0 & 0xFFFFu) << 16;                          // dim0[15:0] at bit48
  g1[2] = (tensor_d0 >> 16) | ((tensor_d1 & 0xFFFFu) << 16);    // dim0 hi | dim1 lo
  g1[3] = (tensor_d1 >> 16) | (tile_d0 << 16);                  // dim1 hi | tile_dim0
  g1[4] = tile_d1;                                              // tile_dim1 (tile_dim2=0)
  g1[5] = (uint32_t)stride0;                                    // dim0_stride lo32
  g1[6] = (uint32_t)((stride0 >> 32) & 0xFFFFu);                // stride hi | dim1_stride=0
  g1[7] = 0u;
  asm volatile("tensor_load_to_lds %0, %1"
               :: "s"(g0), "s"(g1), "v"(lds_ptr) : "memory");
}

// ---- kernel 0: f32 -> f16 weight conversion --------------------------------
__global__ void cvt_f32_to_f16(const float* __restrict__ s, _Float16* __restrict__ d, int n) {
  int i = blockIdx.x * blockDim.x + threadIdx.x;
  if (i < n) d[i] = (_Float16)s[i];
}

// ---- kernel 1: per-head linear projection ----------------------------------
__global__ __launch_bounds__(128) void proj_kernel(const float* __restrict__ X,
                                                   const _Float16* __restrict__ W16,
                                                   const float* __restrict__ bias,
                                                   _Float16* __restrict__ dst,
                                                   int transposeV, float scale) {
  int wave = threadIdx.x >> 5, lane = threadIdx.x & 31;
  int half = lane >> 4, l16 = lane & 15;
  int s0 = (blockIdx.x * 4 + wave) * 16;
  int bh = blockIdx.y;                  // b*HEADS + h
  int b = bh >> 4, h = bh & 15;

  const float* abase = X + ((size_t)b * SEQ + s0) * EMBED + (size_t)h * HDIM;
  v16h a0 = load_a_f32(abase, EMBED, l16, 0, half);
  v16h a1 = load_a_f32(abase, EMBED, l16, 32, half);

  v8f acc[4] = {};
#pragma unroll
  for (int t = 0; t < 4; ++t) {
    v16h b0 = load_b_t(W16, HDIM, t * 16, 0, half, l16);
    acc[t] = wmma_f32_f16(a0, b0, acc[t]);
    v16h b1 = load_b_t(W16, HDIM, t * 16, 32, half, l16);
    acc[t] = wmma_f32_f16(a1, b1, acc[t]);
  }
#pragma unroll
  for (int t = 0; t < 4; ++t) {
    int e = t * 16 + l16;
    float bb = bias[e];
#pragma unroll
    for (int r = 0; r < 8; ++r) {
      int row = r + half * 8;
      float val = (acc[t][r] + bb) * scale;
      if (transposeV)
        dst[((size_t)bh * HDIM + e) * SEQ + (s0 + row)] = (_Float16)val;
      else
        dst[((size_t)bh * SEQ + (s0 + row)) * HDIM + e] = (_Float16)val;
    }
  }
}

// ---- kernel 2: flash attention (one wave = 16 queries x d=64) --------------
// K/V tiles are staged block-wide into LDS by the Tensor Data Mover, double
// buffered, tracked with TENSORcnt.
__global__ __launch_bounds__(128) void attn_kernel(const _Float16* __restrict__ qh,
                                                   const _Float16* __restrict__ kh,
                                                   const _Float16* __restrict__ vT,
                                                   _Float16* __restrict__ attnout) {
  __shared__ _Float16 kbuf[2][32 * 64];   // [key][d]   4KB per buffer
  __shared__ _Float16 vbuf[2][64 * 32];   // [d][key]   4KB per buffer
  __shared__ _Float16 Pt[4][16 * 32];     // per-wave P tile

  int wave = threadIdx.x >> 5, lane = threadIdx.x & 31;
  int half = lane >> 4, l16 = lane & 15;
  int h = blockIdx.y, b = blockIdx.z;
  int bh = b * HEADS + h;
  int q0 = (blockIdx.x * 4 + wave) * 16;

  const _Float16* qhead = qh + (size_t)bh * SEQ * HDIM;
  const _Float16* khead = kh + (size_t)bh * SEQ * HDIM;
  const _Float16* vhead = vT + (size_t)bh * HDIM * SEQ;

  // Q tile (already pre-scaled by 1/sqrt(64) in proj_kernel)
  v16h aq0 = load_a_f16(qhead + (size_t)q0 * HDIM, HDIM, l16, 0, half);
  v16h aq1 = load_a_f16(qhead + (size_t)q0 * HDIM, HDIM, l16, 32, half);

  v8f acc_o[4] = {};
  float mrun[8], lrun[8];
#pragma unroll
  for (int r = 0; r < 8; ++r) { mrun[r] = -1e30f; lrun[r] = 0.0f; }

  _Float16* P = Pt[wave];

  // prologue: TDM-stage chunk 0 (K: 32 rows x 32 dwords; V^T: 64 rows x 16 dwords)
  if (wave == 0) {
    tdm_load_2d(kbuf[0], khead,  32, 32, 32, 32, 32);
    tdm_load_2d(vbuf[0], vhead,  16, 64, 16, 64, (uint64_t)(SEQ / 2));
  }

  int ibuf = 0;
  for (int kc = 0; kc < SEQ; kc += 32, ibuf ^= 1) {
    if (wave == 0) {
      if (kc + 32 < SEQ) {   // prefetch next chunk, then drain current chunk's 2 ops
        tdm_load_2d(kbuf[ibuf ^ 1], khead + (size_t)(kc + 32) * HDIM, 32, 32, 32, 32, 32);
        tdm_load_2d(vbuf[ibuf ^ 1], vhead + (kc + 32),                16, 64, 16, 64,
                    (uint64_t)(SEQ / 2));
        __builtin_amdgcn_s_wait_tensorcnt(2);
      } else {
        __builtin_amdgcn_s_wait_tensorcnt(0);
      }
    }
    __syncthreads();                       // current-chunk tiles visible to all waves

    const _Float16* kb_ = kbuf[ibuf];
    const _Float16* vb_ = vbuf[ibuf];

    // scores: S[16 x 32] = Q(16x64) . K^T(64x32)
    v8f s0 = {}, s1 = {};
    v16h bk;
    bk = load_b_t(kb_, HDIM, 0,  0,  half, l16); s0 = wmma_f32_f16(aq0, bk, s0);
    bk = load_b_t(kb_, HDIM, 0,  32, half, l16); s0 = wmma_f32_f16(aq1, bk, s0);
    bk = load_b_t(kb_, HDIM, 16, 0,  half, l16); s1 = wmma_f32_f16(aq0, bk, s1);
    bk = load_b_t(kb_, HDIM, 16, 32, half, l16); s1 = wmma_f32_f16(aq1, bk, s1);

    // online softmax per row (row = r + 8*half, replicated across 16 lanes)
#pragma unroll
    for (int r = 0; r < 8; ++r) {
      float a = s0[r], c = s1[r];
      float mx   = rowred_max(fmaxf(a, c));
      float mnew = fmaxf(mrun[r], mx);
      float corr = __expf(mrun[r] - mnew);
      float p0 = __expf(a - mnew);
      float p1 = __expf(c - mnew);
      lrun[r] = lrun[r] * corr + rowred_sum(p0 + p1);
      mrun[r] = mnew;
#pragma unroll
      for (int t = 0; t < 4; ++t) acc_o[t][r] *= corr;
      int row = r + half * 8;
      P[row * 32 + l16]      = (_Float16)p0;
      P[row * 32 + 16 + l16] = (_Float16)p1;
    }
    __syncthreads();                       // D-layout -> A-layout via LDS
    v16h ap = load_a_f16(P, 32, l16, 0, half);
    __syncthreads();

    // O += P(16x32) . V(32x64)
#pragma unroll
    for (int t = 0; t < 4; ++t) {
      v16h bv = load_b_t(vb_, 32, t * 16, 0, half, l16);
      acc_o[t] = wmma_f32_f16(ap, bv, acc_o[t]);
    }
    __syncthreads();                       // all reads of ibuf done before TDM reuses it
  }

  // normalize and store to [b][s][h*64+d] f16
#pragma unroll
  for (int r = 0; r < 8; ++r) {
    float inv = 1.0f / lrun[r];
    int row = r + half * 8;
    size_t orow = ((size_t)b * SEQ + q0 + row) * EMBED + (size_t)h * HDIM;
#pragma unroll
    for (int t = 0; t < 4; ++t)
      attnout[orow + t * 16 + l16] = (_Float16)(acc_o[t][r] * inv);
  }
}

// ---- kernel 3: output projection out = attn . Wo^T + bo --------------------
__global__ __launch_bounds__(128) void oproj_kernel(const _Float16* __restrict__ attn,
                                                    const _Float16* __restrict__ Wo16,
                                                    const float* __restrict__ bo,
                                                    float* __restrict__ out) {
  int wave = threadIdx.x >> 5, lane = threadIdx.x & 31;
  int half = lane >> 4, l16 = lane & 15;
  int bs0 = (blockIdx.x * 4 + wave) * 16;   // token tile
  int e0  = blockIdx.y * 64;                // output-column tile

  const _Float16* arow = attn + (size_t)bs0 * EMBED;
  v8f acc[4] = {};
  for (int f = 0; f < EMBED; f += 32) {
    v16h a = load_a_f16(arow, EMBED, l16, f, half);
#pragma unroll
    for (int t = 0; t < 4; ++t) {
      v16h bw = load_b_t(Wo16, EMBED, e0 + t * 16, f, half, l16);
      acc[t] = wmma_f32_f16(a, bw, acc[t]);
    }
  }
#pragma unroll
  for (int t = 0; t < 4; ++t) {
    int e = e0 + t * 16 + l16;
    float bb = bo[e];
#pragma unroll
    for (int r = 0; r < 8; ++r) {
      int row = r + half * 8;
      out[(size_t)(bs0 + row) * EMBED + e] = acc[t][r] + bb;
    }
  }
}

// ---- launcher ---------------------------------------------------------------
extern "C" void kernel_launch(void* const* d_in, const int* in_sizes, int n_in,
                              void* d_out, int out_size, void* d_ws, size_t ws_size,
                              hipStream_t stream) {
  (void)in_sizes; (void)n_in; (void)out_size; (void)ws_size;
  const float* q  = (const float*)d_in[0];
  const float* k  = (const float*)d_in[1];
  const float* v  = (const float*)d_in[2];
  const float* Wq = (const float*)d_in[3];
  const float* bq = (const float*)d_in[4];
  const float* Wk = (const float*)d_in[5];
  const float* bk = (const float*)d_in[6];
  const float* Wv = (const float*)d_in[7];
  const float* bv = (const float*)d_in[8];
  const float* Wo = (const float*)d_in[9];
  const float* bo = (const float*)d_in[10];
  float* out = (float*)d_out;

  // workspace map (all offsets 256B-aligned); total ~69.3 MB
  char* w = (char*)d_ws;
  _Float16* Wq16 = (_Float16*)w; w += (size_t)HDIM * HDIM * 2;
  _Float16* Wk16 = (_Float16*)w; w += (size_t)HDIM * HDIM * 2;
  _Float16* Wv16 = (_Float16*)w; w += (size_t)HDIM * HDIM * 2;
  _Float16* Wo16 = (_Float16*)w; w += (size_t)EMBED * EMBED * 2;
  _Float16* qh   = (_Float16*)w; w += (size_t)BATCH * HEADS * SEQ * HDIM * 2;
  _Float16* kh   = (_Float16*)w; w += (size_t)BATCH * HEADS * SEQ * HDIM * 2;
  _Float16* vT   = (_Float16*)w; w += (size_t)BATCH * HEADS * HDIM * SEQ * 2;
  _Float16* attn = (_Float16*)w; w += (size_t)BATCH * SEQ * EMBED * 2;

  // 0) weight conversion
  cvt_f32_to_f16<<<(HDIM * HDIM + 255) / 256, 256, 0, stream>>>(Wq, Wq16, HDIM * HDIM);
  cvt_f32_to_f16<<<(HDIM * HDIM + 255) / 256, 256, 0, stream>>>(Wk, Wk16, HDIM * HDIM);
  cvt_f32_to_f16<<<(HDIM * HDIM + 255) / 256, 256, 0, stream>>>(Wv, Wv16, HDIM * HDIM);
  cvt_f32_to_f16<<<(EMBED * EMBED + 255) / 256, 256, 0, stream>>>(Wo, Wo16, EMBED * EMBED);

  // 1) per-head projections (Q folded with 1/sqrt(d); V stored transposed)
  dim3 pg(SEQ / 64, BATCH * HEADS);
  proj_kernel<<<pg, 128, 0, stream>>>(q, Wq16, bq, qh, 0, 0.125f);
  proj_kernel<<<pg, 128, 0, stream>>>(k, Wk16, bk, kh, 0, 1.0f);
  proj_kernel<<<pg, 128, 0, stream>>>(v, Wv16, bv, vT, 1, 1.0f);

  // 2) flash attention (TDM-staged K/V, double buffered)
  attn_kernel<<<dim3(SEQ / 64, HEADS, BATCH), 128, 0, stream>>>(qh, kh, vT, attn);

  // 3) output projection
  oproj_kernel<<<dim3(BATCH * SEQ / 64, EMBED / 64), 128, 0, stream>>>(attn, Wo16, bo, out);
}
